// AuxiliaryTransformerBlock_23347442221598
// MI455X (gfx1250) — matrix operand order repaired and checked by reading, not compiled, not run on hardware
//
#include <hip/hip_runtime.h>
#include <hip/hip_bf16.h>
#include <math.h>

// Problem constants
#define BB    256
#define SS    200
#define DDIM  256
#define HH    8
#define DH    32
#define NROWS (BB * SS)   // 51200
#define SPR   208         // S padded for M (rows), multiple of 16
#define SPC   224         // S padded for N/K (cols), multiple of 32

typedef __attribute__((ext_vector_type(16))) _Float16 v16h;
typedef __attribute__((ext_vector_type(8)))  _Float16 v8h;
typedef __attribute__((ext_vector_type(8)))  float    v8f;

// ---------------------------------------------------------------------------
// WMMA fragment loaders (CDNA5 16-bit layouts, wave32)
// A 16x32 (MxK), row-major source with leading dim lda (halves):
//   lane m=lane&15, kh=lane>>4; elems 0..7 = K[kh*8 .. kh*8+7],
//   elems 8..15 = K[16+kh*8 .. 16+kh*8+7]
__device__ __forceinline__ v16h load_a_frag(const _Float16* p, int lda, int lane) {
  const int m = lane & 15, kh = lane >> 4;
  const _Float16* r = p + (size_t)m * lda + kh * 8;
  v8h g0 = *(const v8h*)(r);
  v8h g1 = *(const v8h*)(r + 16);
  v16h a;
#pragma unroll
  for (int i = 0; i < 8; ++i) { a[i] = g0[i]; a[8 + i] = g1[i]; }
  return a;
}

// B 32x16 (KxN) where the source is the N x K row-major matrix (B = src^T):
//   lane n=lane&15, kh=lane>>4; elems 0..7 = src[n][k0 + kh*8 + e],
//   elems 8..15 = src[n][k0 + 16 + kh*8 + (e-8)]
__device__ __forceinline__ v16h load_b_frag(const _Float16* p, int ldb, int lane) {
  const int n = lane & 15, kh = lane >> 4;
  const _Float16* r = p + (size_t)n * ldb + kh * 8;
  v8h g0 = *(const v8h*)(r);
  v8h g1 = *(const v8h*)(r + 16);
  v16h b;
#pragma unroll
  for (int i = 0; i < 8; ++i) { b[i] = g0[i]; b[8 + i] = g1[i]; }
  return b;
}

__device__ __forceinline__ v8f wmma_f16(v16h a, v16h b, v8f c) {
  return __builtin_amdgcn_wmma_f32_16x16x32_f16(false, a, false, b, (short)0, c,
                                                false, false);
}

// ---------------------------------------------------------------------------
// Block-cooperative GEMM tile: block = 256 threads = 8 waves covering
// 2 M-tiles x 4 column groups (32 rows x 256 cols). The 256x32 weight k-chunk
// (16 KB f16) is staged into LDS, double-buffered (32 KB), so B-fragments come
// from ds_load while A-fragments stream from global; staging of chunk kt+1
// overlaps the WMMAs of chunk kt.
//   wbuf: __shared__ _Float16[2 * 8192]
__device__ __forceinline__ void gemm_tile_lds(const _Float16* __restrict__ X,
                                              const _Float16* __restrict__ W,
                                              _Float16* wbuf, int row0, int col0,
                                              int lane, int tid, v8f acc[4]) {
#pragma unroll
  for (int j = 0; j < 4; ++j) acc[j] = (v8f){};

  // stage chunk 0: thread t copies W[t][0..31]
  {
    const v8h* s = (const v8h*)(W + (size_t)tid * DDIM);
    v8h* d = (v8h*)(wbuf + tid * 32);
    d[0] = s[0]; d[1] = s[1]; d[2] = s[2]; d[3] = s[3];
  }
  __syncthreads();

  for (int kt = 0; kt < 8; ++kt) {
    _Float16* cur = wbuf + (kt & 1) * 8192;
    if (kt < 7) {  // stage next chunk while computing on current
      const v8h* s = (const v8h*)(W + (size_t)tid * DDIM + (kt + 1) * 32);
      v8h* d = (v8h*)(wbuf + ((kt + 1) & 1) * 8192 + tid * 32);
      d[0] = s[0]; d[1] = s[1]; d[2] = s[2]; d[3] = s[3];
    }
    v16h a = load_a_frag(X + (size_t)row0 * DDIM + kt * 32, DDIM, lane);
#pragma unroll
    for (int j = 0; j < 4; ++j) {
      v16h b = load_b_frag(cur + (col0 + j * 16) * 32, 32, lane);
      acc[j] = wmma_f16(a, b, acc[j]);
    }
    __syncthreads();
  }
}

// ---------------------------------------------------------------------------
// Weight f32 -> f16 conversion
__global__ void wcvt_kernel(const float* __restrict__ w,
                            _Float16* __restrict__ wh, int n) {
  int i = blockIdx.x * blockDim.x + threadIdx.x;
  if (i < n) wh[i] = (_Float16)w[i];
}

// LayerNorm (torch variant: unbiased std, eps added to std), wave-per-row,
// f32 in -> f16 out. 256 threads = 8 waves = 8 rows per block.
__global__ __launch_bounds__(256) void ln_kernel(const float* __restrict__ x,
                                                 const float* __restrict__ g,
                                                 const float* __restrict__ bta,
                                                 _Float16* __restrict__ y,
                                                 int nrows) {
  const int lane = threadIdx.x & 31, wave = threadIdx.x >> 5;
  const int row = blockIdx.x * 8 + wave;
  if (row >= nrows) return;
  const float* xr = x + (size_t)row * DDIM;
  float v[8], s = 0.f, s2 = 0.f;
#pragma unroll
  for (int i = 0; i < 8; ++i) {
    v[i] = xr[lane * 8 + i];
    s += v[i];
    s2 += v[i] * v[i];
  }
#pragma unroll
  for (int off = 16; off; off >>= 1) {
    s  += __shfl_xor(s,  off, 32);
    s2 += __shfl_xor(s2, off, 32);
  }
  const float mean = s * (1.f / 256.f);
  const float var  = (s2 - 256.f * mean * mean) * (1.f / 255.f);
  const float inv  = 1.f / (sqrtf(var) + 1e-6f);
  _Float16* yr = y + (size_t)row * DDIM;
#pragma unroll
  for (int i = 0; i < 8; ++i) {
    const int c = lane * 8 + i;
    yr[c] = (_Float16)(g[c] * (v[i] - mean) * inv + bta[c]);
  }
}

// kk = img_n + title_n @ Wb^T + bb   (f16 out)
__global__ __launch_bounds__(256) void gemm_kb_kernel(
    const _Float16* __restrict__ Xh, const _Float16* __restrict__ Wh,
    const float* __restrict__ bias, const _Float16* __restrict__ imgn,
    _Float16* __restrict__ kk) {
  __shared__ _Float16 wbuf[2 * 8192];
  const int tid = threadIdx.x, lane = tid & 31, wave = tid >> 5;
  const int row0 = (blockIdx.x * 2 + (wave >> 2)) * 16;
  const int col0 = (wave & 3) * 64;
  v8f acc[4];
  gemm_tile_lds(Xh, Wh, wbuf, row0, col0, lane, tid, acc);
#pragma unroll
  for (int j = 0; j < 4; ++j) {
    const int n = col0 + j * 16 + (lane & 15);
    const float bn = bias[n];
#pragma unroll
    for (int i = 0; i < 8; ++i) {
      const size_t m = (size_t)row0 + (lane >> 4) * 8 + i;
      const size_t o = m * DDIM + n;
      kk[o] = (_Float16)((float)imgn[o] + acc[j][i] + bn);
    }
  }
}

// h1 = hidden_img + ctx @ Wo^T + bo   (f32 out)
__global__ __launch_bounds__(256) void gemm_wo_kernel(
    const _Float16* __restrict__ ctx, const _Float16* __restrict__ Wh,
    const float* __restrict__ bias, const float* __restrict__ hid,
    float* __restrict__ h1) {
  __shared__ _Float16 wbuf[2 * 8192];
  const int tid = threadIdx.x, lane = tid & 31, wave = tid >> 5;
  const int row0 = (blockIdx.x * 2 + (wave >> 2)) * 16;
  const int col0 = (wave & 3) * 64;
  v8f acc[4];
  gemm_tile_lds(ctx, Wh, wbuf, row0, col0, lane, tid, acc);
#pragma unroll
  for (int j = 0; j < 4; ++j) {
    const int n = col0 + j * 16 + (lane & 15);
    const float bn = bias[n];
#pragma unroll
    for (int i = 0; i < 8; ++i) {
      const size_t m = (size_t)row0 + (lane >> 4) * 8 + i;
      const size_t o = m * DDIM + n;
      h1[o] = hid[o] + acc[j][i] + bn;
    }
  }
}

// out = h1 + gelu_tanh(scale_ffn[s] * (h1n @ W1^T + b1))
__global__ __launch_bounds__(256) void gemm_ffn_kernel(
    const _Float16* __restrict__ h1n, const _Float16* __restrict__ Wh,
    const float* __restrict__ bias, const float* __restrict__ scale_ffn,
    const float* __restrict__ h1, float* __restrict__ out) {
  __shared__ _Float16 wbuf[2 * 8192];
  const int tid = threadIdx.x, lane = tid & 31, wave = tid >> 5;
  const int row0 = (blockIdx.x * 2 + (wave >> 2)) * 16;
  const int col0 = (wave & 3) * 64;
  v8f acc[4];
  gemm_tile_lds(h1n, Wh, wbuf, row0, col0, lane, tid, acc);
  const float GC = 0.7978845608028654f;
#pragma unroll
  for (int j = 0; j < 4; ++j) {
    const int n = col0 + j * 16 + (lane & 15);
    const float bn = bias[n];
#pragma unroll
    for (int i = 0; i < 8; ++i) {
      const int m = row0 + (lane >> 4) * 8 + i;
      const int s = m % SS;
      const float z = scale_ffn[s] * (acc[j][i] + bn);
      const float g = 0.5f * z * (1.f + tanhf(GC * (z + 0.044715f * z * z * z)));
      const size_t o = (size_t)m * DDIM + n;
      out[o] = h1[o] + g;
    }
  }
}

// ---------------------------------------------------------------------------
// Attention: one workgroup per (b,h); everything resident in LDS (~322 KB).
// LDS layout (byte offsets, all 16B-aligned):
//   qs  [208][32] f16 @ 0       (13312 B)   Q rows, zero-padded
//   ks  [224][32] f16 @ 13312   (14336 B)   K (= kk head slice), zero-padded
//   vt  [32][224] f16 @ 27648   (14336 B)   V transposed (d-major)
//   sc  [208][224] f32 @ 41984  (186368 B)  scores
//   ph  [208][224] f16 @ 228352 (93184 B)   probabilities (f16 for WMMA)
//   mk  [224] int @ 321536      (896 B)     key mask
__global__ __launch_bounds__(256) void attn_kernel(
    const _Float16* __restrict__ imgn, const _Float16* __restrict__ kk,
    const int* __restrict__ mask, const float* __restrict__ scale_attn,
    _Float16* __restrict__ ctx) {
  extern __shared__ char smem[];
  _Float16* qs = (_Float16*)(smem);
  _Float16* ks = (_Float16*)(smem + 13312);
  _Float16* vt = (_Float16*)(smem + 27648);
  float*    sc = (float*)(smem + 41984);
  _Float16* ph = (_Float16*)(smem + 228352);
  int*      mk = (int*)(smem + 321536);

  const int b = blockIdx.x >> 3;
  const int h = blockIdx.x & 7;
  const int tid = threadIdx.x;
  const int lane = tid & 31, wave = tid >> 5;

  // Phase 1: stage Q/K/V^T into LDS (zero-pad rows >= 200)
  for (int idx = tid; idx < SPC * DH; idx += 256) {
    const int s = idx >> 5, d = idx & 31;
    _Float16 q = (_Float16)0.f, k = (_Float16)0.f;
    if (s < SS) {
      const size_t g = ((size_t)(b * SS + s)) * DDIM + h * DH + d;
      q = imgn[g];
      k = kk[g];
    }
    if (s < SPR) qs[s * DH + d] = q;
    ks[idx] = k;
    vt[d * SPC + s] = q;  // V == Q (img_n)
  }
  for (int s = tid; s < SPC; s += 256) mk[s] = (s < SS) ? mask[b * SS + s] : 0;
  __syncthreads();

  // Phase 2: scores = (Q @ K^T) / sqrt(32) * scale_attn[h,q], masked
  const float rsc = 0.17677669529663687f;  // 1/sqrt(32)
  for (int t = wave; t < 13 * 14; t += 8) {  // wave-uniform trip count
    const int it = t / 14, jt = t % 14;
    v16h a = load_a_frag(qs + it * 16 * DH, DH, lane);
    v16h bf = load_b_frag(ks + jt * 16 * DH, DH, lane);
    v8f acc = (v8f){};
    acc = wmma_f16(a, bf, acc);
    const int col = jt * 16 + (lane & 15);
    const bool dead = (col >= SS) || (mk[col] == 0);
#pragma unroll
    for (int i = 0; i < 8; ++i) {
      const int row = it * 16 + (lane >> 4) * 8 + i;
      const int qi = (row < SS) ? row : (SS - 1);
      const float v = acc[i] * rsc * scale_attn[h * SS + qi];
      sc[row * SPC + col] = dead ? -1e9f : v;
    }
  }
  __syncthreads();

  // Phase 3: row softmax (224 cols = exactly 7 per lane), write f16 probs
  for (int row = wave; row < SPR; row += 8) {
    const float* r = sc + row * SPC;
    float mx = -3.4e38f;
#pragma unroll
    for (int c = lane; c < SPC; c += 32) mx = fmaxf(mx, r[c]);
#pragma unroll
    for (int off = 16; off; off >>= 1) mx = fmaxf(mx, __shfl_xor(mx, off, 32));
    float sum = 0.f;
#pragma unroll
    for (int c = lane; c < SPC; c += 32) sum += __expf(r[c] - mx);
#pragma unroll
    for (int off = 16; off; off >>= 1) sum += __shfl_xor(sum, off, 32);
    const float inv = 1.f / sum;
    _Float16* pr = ph + row * SPC;
#pragma unroll
    for (int c = lane; c < SPC; c += 32)
      pr[c] = (_Float16)(__expf(r[c] - mx) * inv);
  }
  __syncthreads();

  // Phase 4: ctx = prob @ V  (13 m-tiles x 2 n-tiles, K = 224 = 7 steps)
  for (int t = wave; t < 26; t += 8) {
    const int it = t >> 1, jt = t & 1;
    v8f acc = (v8f){};
    for (int kt = 0; kt < 7; ++kt) {
      v16h a  = load_a_frag(ph + it * 16 * SPC + kt * 32, SPC, lane);
      v16h bf = load_b_frag(vt + jt * 16 * SPC + kt * 32, SPC, lane);
      acc = wmma_f16(a, bf, acc);
    }
    const int d = jt * 16 + (lane & 15);
#pragma unroll
    for (int i = 0; i < 8; ++i) {
      const int q = it * 16 + (lane >> 4) * 8 + i;
      if (q < SS)
        ctx[((size_t)(b * SS + q)) * DDIM + h * DH + d] = (_Float16)acc[i];
    }
  }
}

// ---------------------------------------------------------------------------
extern "C" void kernel_launch(void* const* d_in, const int* in_sizes, int n_in,
                              void* d_out, int out_size, void* d_ws,
                              size_t ws_size, hipStream_t stream) {
  const float* hidden_img   = (const float*)d_in[0];
  const float* hidden_title = (const float*)d_in[1];
  const int*   mask         = (const int*)d_in[2];
  const float* a_img   = (const float*)d_in[3];
  const float* b_img   = (const float*)d_in[4];
  const float* a_title = (const float*)d_in[5];
  const float* b_title = (const float*)d_in[6];
  const float* Wb = (const float*)d_in[7];
  const float* bb = (const float*)d_in[8];
  const float* Wo = (const float*)d_in[9];
  const float* bo = (const float*)d_in[10];
  const float* scale_attn = (const float*)d_in[11];
  const float* a_out = (const float*)d_in[12];
  const float* b_out = (const float*)d_in[13];
  const float* W1 = (const float*)d_in[14];
  const float* b1 = (const float*)d_in[15];
  const float* scale_ffn = (const float*)d_in[16];
  float* out = (float*)d_out;

  // Workspace layout (bytes). NROWS*DDIM f16 = 26,214,400 B each.
  char* ws = (char*)d_ws;
  _Float16* imgn     = (_Float16*)(ws + 0);           // img_n (Q,V)
  _Float16* titleCtx = (_Float16*)(ws + 26214400);    // title_n, later ctx
  _Float16* kkH1n    = (_Float16*)(ws + 52428800);    // kk, later h1_n
  float*    h1       = (float*)(ws + 78643200);       // h1 f32 (52,428,800 B)
  _Float16* WbH      = (_Float16*)(ws + 131072000);
  _Float16* WoH      = (_Float16*)(ws + 131203072);
  _Float16* W1H      = (_Float16*)(ws + 131334144);

  wcvt_kernel<<<256, 256, 0, stream>>>(Wb, WbH, DDIM * DDIM);
  wcvt_kernel<<<256, 256, 0, stream>>>(Wo, WoH, DDIM * DDIM);
  wcvt_kernel<<<256, 256, 0, stream>>>(W1, W1H, DDIM * DDIM);

  ln_kernel<<<NROWS / 8, 256, 0, stream>>>(hidden_img, a_img, b_img, imgn, NROWS);
  ln_kernel<<<NROWS / 8, 256, 0, stream>>>(hidden_title, a_title, b_title,
                                           titleCtx, NROWS);

  // kk = img_n + title_n @ Wb^T + bb  (since q.k + q.k_b == q.(k + k_b))
  gemm_kb_kernel<<<NROWS / 32, 256, 0, stream>>>(titleCtx, WbH, bb, imgn, kkH1n);

  // attention, one WG per (b,h), 322,432 B dynamic LDS
  attn_kernel<<<BB * HH, 256, 322432, stream>>>(imgn, kkH1n, mask, scale_attn,
                                                titleCtx);

  // h1 = hidden_img + ctx @ Wo^T + bo
  gemm_wo_kernel<<<NROWS / 32, 256, 0, stream>>>(titleCtx, WoH, bo, hidden_img, h1);

  // h1_n = LN(h1)
  ln_kernel<<<NROWS / 8, 256, 0, stream>>>(h1, a_out, b_out, kkH1n, NROWS);

  // out = h1 + gelu(scale_ffn * (h1_n @ W1^T + b1))
  gemm_ffn_kernel<<<NROWS / 32, 256, 0, stream>>>(kkH1n, W1H, b1, scale_ffn, h1,
                                                  out);
}
